// MACEModel_78683800863183
// MI455X (gfx1250) — matrix operand order repaired and checked by reading, not compiled, not run on hardware
//
#include <hip/hip_runtime.h>
#include <hip/hip_bf16.h>
#include <math.h>
#include <stdint.h>

typedef __attribute__((ext_vector_type(16))) _Float16 v16h;
typedef __attribute__((ext_vector_type(8)))  _Float16 v8h;
typedef __attribute__((ext_vector_type(4)))  _Float16 v4h;
typedef __attribute__((ext_vector_type(8)))  float    v8f;
typedef int v4i __attribute__((vector_size(16)));   // matches builtin param type from hipcc diagnostic

// ---------------- gfx1250 async-to-LDS support (compile-safe gating) ----------------
#if defined(__has_builtin)
#  if __has_builtin(__builtin_amdgcn_global_load_async_to_lds_b128)
#    define HAVE_ASYNC_LDS 1
#  endif
#endif
#ifndef HAVE_ASYNC_LDS
#  define HAVE_ASYNC_LDS 0
#endif

#if HAVE_ASYNC_LDS
__device__ inline void async_g2l_b128(const void* g, void* l) {
  __builtin_amdgcn_global_load_async_to_lds_b128((v4i*)g, (v4i*)l, 0, 0);
}
__device__ inline void async_wait0() {
#if __has_builtin(__builtin_amdgcn_s_wait_asynccnt)
  __builtin_amdgcn_s_wait_asynccnt(0);
#else
  asm volatile("s_wait_asynccnt 0x0" ::: "memory");
#endif
}
#endif

// ---------------- constants ----------------
__constant__ int LOFF[4] = {0, 1, 4, 9};          // slot offset of irrep l in the 16-slot layout
__constant__ int TRIPS[23][3] = {
  {0,0,0},{0,1,1},{0,2,2},{0,3,3},
  {1,0,1},{1,1,0},{1,1,2},{1,2,1},{1,2,3},{1,3,2},
  {2,0,2},{2,1,1},{2,1,3},{2,2,0},{2,2,2},{2,3,1},{2,3,3},
  {3,0,3},{3,1,2},{3,2,1},{3,2,3},{3,3,0},{3,3,2}};
__constant__ int TRIP_OFF[23] = {0,1,10,35, 84,93,102,147,192,297,
                                 402,427,472,577,602,727,832,
                                 1077,1126,1231,1336,1581,1630};
// packed CG base offset indexed by l1*16+l2*4+l3 (-1 = invalid triple)
__constant__ int CGOFF64[64] = {
     0,  -1,  -1,  -1,  -1,   1,  -1,  -1,  -1,  -1,  10,  -1,  -1,  -1,  -1,  35,
    -1,  84,  -1,  -1,  93,  -1, 102,  -1,  -1, 147,  -1, 192,  -1,  -1, 297,  -1,
    -1,  -1, 402,  -1,  -1, 427,  -1, 472, 577,  -1, 602,  -1,  -1, 727,  -1, 832,
    -1,  -1,  -1,1077,  -1,  -1,1126,  -1,  -1,1231,  -1,1336,1581,  -1,1630,  -1};
__constant__ int PATHS0[4][3]  = {{0,0,0},{0,1,1},{0,2,2},{0,3,3}};
__constant__ int PATHS1[10][3] = {{0,0,0},{0,1,1},{0,2,2},{0,3,3},
                                  {1,0,1},{1,1,0},{1,1,2},{1,2,1},{1,2,3},{1,3,2}};
#define CGN 1875

// ---------------- device complex helpers for CG init ----------------
struct dcx { double re, im; };
__device__ inline dcx cmul(dcx a, dcx b){ return {a.re*b.re - a.im*b.im, a.re*b.im + a.im*b.re}; }
__device__ inline dcx cadd(dcx a, dcx b){ return {a.re+b.re, a.im+b.im}; }
__device__ inline dcx cconj(dcx a){ return {a.re, -a.im}; }
__device__ inline double dfact(int n){ double f = 1.0; for (int i = 2; i <= n; ++i) f *= (double)i; return f; }

__device__ double su2cg(int j1, int j2, int j3, int m1, int m2, int m3) {
  if (m3 != m1 + m2) return 0.0;
  int vmin = 0;
  if (j2 - j3 - m1 > vmin) vmin = j2 - j3 - m1;
  if (j1 - j3 + m2 > vmin) vmin = j1 - j3 + m2;
  int vmax = j1 + j2 - j3;
  if (j1 - m1 < vmax) vmax = j1 - m1;
  if (j2 + m2 < vmax) vmax = j2 + m2;
  if (vmax < vmin) return 0.0;
  double pref = sqrt((2.0*j3 + 1.0) * dfact(j1+j2-j3) * dfact(j1-j2+j3) * dfact(-j1+j2+j3) / dfact(j1+j2+j3+1)
                     * dfact(j1+m1) * dfact(j1-m1) * dfact(j2+m2) * dfact(j2-m2) * dfact(j3+m3) * dfact(j3-m3));
  double s = 0.0;
  for (int v = vmin; v <= vmax; ++v) {
    double d = dfact(v)*dfact(j1+j2-j3-v)*dfact(j1-m1-v)*dfact(j2+m2-v)*dfact(j3-j2+m1+v)*dfact(j3-j1-m2+v);
    s += ((v & 1) ? -1.0 : 1.0) / d;
  }
  return pref * s;
}

// q[l] entry [a][b] (a,b in 0..2l), including the (-i)^l phase
__device__ dcx qent(int l, int a, int b) {
  int m = a - l;
  const double inv = 0.7071067811865475244;
  dcx v = {0.0, 0.0};
  if (m < 0) {
    if (b == l - m)      v = { inv, 0.0 };
    else if (b == l + m) v = { 0.0, -inv };
  } else if (m == 0) {
    if (b == l) v = { 1.0, 0.0 };
  } else {
    double sg = (m & 1) ? -1.0 : 1.0;
    if (b == l + m)      v = { sg * inv, 0.0 };
    else if (b == l - m) v = { 0.0, sg * inv };
  }
  dcx ph;
  switch (l & 3) { case 0: ph = {1,0}; break; case 1: ph = {0,-1}; break;
                   case 2: ph = {-1,0}; break; default: ph = {0,1}; }
  return cmul(ph, v);
}

__global__ void cg_init_kernel(float* __restrict__ cg) {
  int t = threadIdx.x;
  if (t >= 23) return;
  int l1 = TRIPS[t][0], l2 = TRIPS[t][1], l3 = TRIPS[t][2];
  int n1 = 2*l1+1, n2 = 2*l2+1, n3 = 2*l3+1;
  float* out = cg + TRIP_OFF[t];
  for (int i = 0; i < n1; ++i)
    for (int k = 0; k < n2; ++k)
      for (int m = 0; m < n3; ++m) {
        dcx acc = {0.0, 0.0};
        for (int j = 0; j < n1; ++j)
          for (int ll = 0; ll < n2; ++ll)
            for (int n = 0; n < n3; ++n) {
              double cc = su2cg(l1, l2, l3, j - l1, ll - l2, n - l3);
              if (cc == 0.0) continue;
              dcx q = cmul(cmul(qent(l1, i, j), qent(l2, k, ll)), cconj(qent(l3, m, n)));
              acc = cadd(acc, { q.re * cc, q.im * cc });
            }
        out[(i*n2 + k)*n3 + m] = (float)acc.re;
      }
}

// ---------------- small utility kernels ----------------
__global__ void zero_kernel(float* __restrict__ p, size_t n) {
  size_t i = (size_t)blockIdx.x * 256 + threadIdx.x;
  if (i < n) p[i] = 0.f;
}

__global__ __launch_bounds__(128) void init_feats_kernel(float* __restrict__ fA,
    const float* __restrict__ emb, const int* __restrict__ species, int NN) {
  int n = blockIdx.x, c = threadIdx.x;
  int sp = species[n];
  for (int s = 0; s < 16; ++s)
    fA[((size_t)n*16 + s)*128 + c] = (s == 0) ? emb[(size_t)sp*128 + c] * 0.125f : 0.f;
}

// ---------------- edge geometry: radial basis + spherical harmonics ----------------
__global__ __launch_bounds__(256) void edge_geom_kernel(const float* __restrict__ vecs,
    float* __restrict__ rb, float* __restrict__ Yb, const float* __restrict__ cg, int NE) {
  int e = blockIdx.x * 256 + threadIdx.x;
  if (e >= NE) return;
  float vx = vecs[(size_t)e*3+0], vy = vecs[(size_t)e*3+1], vz = vecs[(size_t)e*3+2];
  float r = sqrtf(vx*vx + vy*vy + vz*vz);
  float ir = 1.f / r;
  float u[3] = { vx*ir, vy*ir, vz*ir };
  float xs = r * 0.2f, xs2 = xs*xs, xp = xs2*xs2*xs;
  float env = 1.f - 21.f*xp + 35.f*xp*xs - 15.f*xp*xs2;
  if (r >= 5.f) env = 0.f;
  const float s25 = 0.632455532f;       // sqrt(2/5)
  for (int j = 0; j < 8; ++j) {
    float b = (float)(j+1) * 0.6283185307f;   // pi/5
    rb[(size_t)e*8 + j] = s25 * sinf(b * r) * ir * env;
  }
  float* yp = Yb + (size_t)e*16;
  yp[0] = 1.f; yp[1] = u[0]; yp[2] = u[1]; yp[3] = u[2];
  const float* c2 = cg + 102;   // (1,1,2)
  const float* c3 = cg + 472;   // (2,1,3)
  float y2[5];
  for (int k = 0; k < 5; ++k) {
    float s = 0.f;
    for (int i = 0; i < 3; ++i)
      for (int j = 0; j < 3; ++j) s += u[i]*u[j]*c2[(i*3+j)*5 + k];
    y2[k] = s; yp[4+k] = s;
  }
  for (int k = 0; k < 7; ++k) {
    float s = 0.f;
    for (int i = 0; i < 5; ++i)
      for (int j = 0; j < 3; ++j) s += y2[i]*u[j]*c3[(i*3+j)*7 + k];
    yp[9+k] = s;
  }
}

// ---------------- WMMA GEMM: C = act((A@B)*scale), f16 frags, f32 accum ----------------
// BM=128, BN=64, BK=32; 8 waves; each wave computes a 32x32 tile (2x2 WMMA).
__device__ inline size_t rmap(int r, int rpg, int gstr, int roff) {
  return rpg ? ((size_t)(r / rpg) * gstr + roff + (r % rpg)) : (size_t)r;
}

__global__ __launch_bounds__(256) void gemm_wmma_kernel(
    const float* __restrict__ A, const float* __restrict__ B, float* __restrict__ Cm,
    int M, int N, int K, float scale, int act, int accum,
    int rpg, int gstr, int roff) {
  __shared__ alignas(16) _Float16 sA[128][48];   // fragment-ready: lane run [hi*8], [16+hi*8]
  __shared__ alignas(16) _Float16 sBt[64][48];   // B transposed: lane run [hi*8], [16+hi*8]
  int tid = threadIdx.x;
  int lane = tid & 31, w = tid >> 5;
  int bM = blockIdx.y * 128, bN = blockIdx.x * 64;
  int wm = (w & 3) * 32, wn = (w >> 2) * 32;
  int l16 = lane & 15, hi = lane >> 4;

  const v8f vz = {0.f,0.f,0.f,0.f,0.f,0.f,0.f,0.f};
  v8f accs[2][2];
  accs[0][0] = vz; accs[0][1] = vz; accs[1][0] = vz; accs[1][1] = vz;

  // per-thread staging assignments (fixed across K loop)
  int arow = tid >> 1;             // 0..127  (A tile row)
  int acb  = (tid & 1) * 16;       // A col base: 0 or 16
  int gr   = bM + arow;
  bool arv = gr < M;
  size_t abase = arv ? rmap(gr, rpg, gstr, roff) * (size_t)K : 0;
  int bkk = tid >> 3;              // 0..31   (B tile k-row)
  int bn0 = (tid & 7) * 8;         // B col base
  bool bnfull = (bN + 64 <= N);

  for (int k0 = 0; k0 < K; k0 += 32) {
    bool kfull = (k0 + 32 <= K);
    // stage A tile (128x32 f32 -> f16), vectorized
    if (kfull && arv) {            // fast path: no per-quad guards
#pragma unroll
      for (int q = 0; q < 4; ++q) {
        float4 v = *(const float4*)&A[abase + k0 + acb + q*4];
        v4h h; h[0]=(_Float16)v.x; h[1]=(_Float16)v.y; h[2]=(_Float16)v.z; h[3]=(_Float16)v.w;
        *(v4h*)&sA[arow][acb + q*4] = h;
      }
    } else {
#pragma unroll
      for (int q = 0; q < 4; ++q) {
        int gk = k0 + acb + q*4;
        float4 v = {0.f,0.f,0.f,0.f};
        if (arv && gk + 3 < K) v = *(const float4*)&A[abase + gk];
        v4h h; h[0]=(_Float16)v.x; h[1]=(_Float16)v.y; h[2]=(_Float16)v.z; h[3]=(_Float16)v.w;
        *(v4h*)&sA[arow][acb + q*4] = h;
      }
    }
    // stage B tile transposed (32x64 f32 -> sBt[n][k] f16)
    if (kfull && bnfull) {
      const float* bp = &B[(size_t)(k0 + bkk) * N + bN + bn0];
#pragma unroll
      for (int q = 0; q < 2; ++q) {
        float4 v = *(const float4*)(bp + q*4);
        sBt[bn0 + q*4 + 0][bkk] = (_Float16)v.x;
        sBt[bn0 + q*4 + 1][bkk] = (_Float16)v.y;
        sBt[bn0 + q*4 + 2][bkk] = (_Float16)v.z;
        sBt[bn0 + q*4 + 3][bkk] = (_Float16)v.w;
      }
    } else {
#pragma unroll
      for (int q = 0; q < 2; ++q) {
        int gk = k0 + bkk;
        int gn = bN + bn0 + q*4;
        float4 v = {0.f,0.f,0.f,0.f};
        if (gk < K && gn + 3 < N) v = *(const float4*)&B[(size_t)gk * N + gn];
        sBt[bn0 + q*4 + 0][bkk] = (_Float16)v.x;
        sBt[bn0 + q*4 + 1][bkk] = (_Float16)v.y;
        sBt[bn0 + q*4 + 2][bkk] = (_Float16)v.z;
        sBt[bn0 + q*4 + 3][bkk] = (_Float16)v.w;
      }
    }
    if (k0 + 32 < K && arv)
      __builtin_prefetch(&A[abase + k0 + 32 + acb], 0, 1);   // global_prefetch_b8
    __syncthreads();

    // build fragments with vector LDS loads
    v16h afr[2], bfr[2];
#pragma unroll
    for (int mt = 0; mt < 2; ++mt) {
      int r = wm + mt*16 + l16;
      v8h lo = *(const v8h*)&sA[r][hi*8];
      v8h hi8 = *(const v8h*)&sA[r][16 + hi*8];
      afr[mt] = __builtin_shufflevector(lo, hi8, 0,1,2,3,4,5,6,7,8,9,10,11,12,13,14,15);
    }
#pragma unroll
    for (int nt = 0; nt < 2; ++nt) {
      int r = wn + nt*16 + l16;
      v8h lo = *(const v8h*)&sBt[r][hi*8];
      v8h hi8 = *(const v8h*)&sBt[r][16 + hi*8];
      bfr[nt] = __builtin_shufflevector(lo, hi8, 0,1,2,3,4,5,6,7,8,9,10,11,12,13,14,15);
    }
#pragma unroll
    for (int mt = 0; mt < 2; ++mt)
#pragma unroll
      for (int nt = 0; nt < 2; ++nt)
        accs[mt][nt] = __builtin_amdgcn_wmma_f32_16x16x32_f16(
            false, afr[mt], false, bfr[nt], (short)0, accs[mt][nt], false, false);
    __syncthreads();
  }

#pragma unroll
  for (int mt = 0; mt < 2; ++mt)
#pragma unroll
    for (int nt = 0; nt < 2; ++nt) {
      v8f a = accs[mt][nt];
#pragma unroll
      for (int vv = 0; vv < 8; ++vv) {
        int row = bM + wm + mt*16 + vv + (hi << 3);
        int col = bN + wn + nt*16 + l16;
        if (row < M && col < N) {
          float x = a[vv] * scale;
          if (act) x = x * (1.f / (1.f + __expf(-x)));
          size_t off = rmap(row, rpg, gstr, roff) * (size_t)N + col;
          if (accum) Cm[off] += x; else Cm[off] = x;
        }
      }
    }
}

// ---------------- edge tensor product + scatter (atomics) ----------------
__global__ __launch_bounds__(128) void tp_scatter_kernel(
    const float* __restrict__ U, const float* __restrict__ Yb, const float* __restrict__ wb,
    const float* __restrict__ cg, const int* __restrict__ senders, const int* __restrict__ receivers,
    float* __restrict__ nf, float* __restrict__ gY, float* __restrict__ gs,
    int e0, int ce, int layer, int wstride) {
  int eb = blockIdx.x;
  if (eb >= ce) return;
  int e = e0 + eb;
  int c = threadIdx.x;
  __shared__ alignas(16) float Ysh[16];
  __shared__ float sred[17];
#if HAVE_ASYNC_LDS
  if (c < 4) async_g2l_b128(&Yb[(size_t)e*16 + c*4], &Ysh[c*4]);  // 64B row -> LDS, async
  if (c < 17) sred[c] = 0.f;
  async_wait0();
#else
  if (c < 16) Ysh[c] = Yb[(size_t)e*16 + c];
  if (c < 17) sred[c] = 0.f;
#endif
  __syncthreads();

  int snd = senders[e], rcv = receivers[e];
  float m0 = U[((size_t)snd*16 + 0)*128 + c];
  float m1[3] = {0.f, 0.f, 0.f};
  if (layer) {
    m1[0] = U[((size_t)snd*16 + 1)*128 + c];
    m1[1] = U[((size_t)snd*16 + 2)*128 + c];
    m1[2] = U[((size_t)snd*16 + 3)*128 + c];
  }
  const float* wr = wb + (size_t)eb * wstride + c;
  const int (*PT)[3] = layer ? PATHS1 : PATHS0;
  int np = layer ? 10 : 4;

  float out[16], gpart[16];
  for (int s = 0; s < 16; ++s) { out[s] = 0.f; gpart[s] = 0.f; }
  float gsp = 0.f;

  for (int p = 0; p < np; ++p) {
    int l1 = PT[p][0], l2 = PT[p][1], l3 = PT[p][2];
    float wpc = wr[(size_t)p * 128];
    int n1 = 2*l1+1, n2 = 2*l2+1, n3 = 2*l3+1;
    const float* cgp = cg + CGOFF64[l1*16 + l2*4 + l3];
    int o2 = LOFF[l2], o3 = LOFF[l3];
    for (int i = 0; i < n1; ++i) {
      float mw = ((l1 == 0) ? m0 : m1[i]) * wpc;
      for (int j = 0; j < n2; ++j) {
        float yj = Ysh[o2 + j];
        for (int k = 0; k < n3; ++k) {
          float cv = cgp[(i*n2 + j)*n3 + k];
          out[o3 + k] += mw * yj * cv;
          gpart[o2 + j] += mw * cv;
          gsp += mw * yj * cv;
        }
      }
    }
  }
  float* nfr = nf + ((size_t)rcv*16)*128 + c;
  for (int s = 0; s < 16; ++s) atomicAdd(nfr + (size_t)s*128, out[s]);
  for (int s = 0; s < 16; ++s) if (gpart[s] != 0.f) atomicAdd(&sred[s], gpart[s]);
  atomicAdd(&sred[16], gsp);
  __syncthreads();
  if (c < 16) gY[(size_t)e*16 + c] += sred[c];
  if (c == 0) gs[e] += sred[16];
}

// ---------------- species-indexed batched matvec (skip connections) ----------------
__global__ __launch_bounds__(128) void species_mm_kernel(
    const float* __restrict__ in, float* __restrict__ out, const float* __restrict__ Wsp,
    const int* __restrict__ species, float scale, int loff, int nm, int accum) {
  int b = blockIdx.x;
  int n = b / nm, mi = loff + b % nm, d = threadIdx.x;
  __shared__ float xv[128];
  xv[d] = in[((size_t)n*16 + mi)*128 + d];
  __syncthreads();
  const float* Wp = Wsp + (size_t)species[n]*16384 + d;
  float acc = 0.f;
  for (int c = 0; c < 128; ++c) acc += xv[c] * Wp[(size_t)c*128];
  acc *= scale;
  float* op = &out[((size_t)n*16 + mi)*128 + d];
  if (accum) *op += acc; else *op = acc;
}

// ---------------- symmetric contraction (correlation 3) ----------------
__device__ inline void cg_contract(const float* a, const float* b, float* o, const float* cg) {
  for (int t = 0; t < 23; ++t) {
    int l1 = TRIPS[t][0], l2 = TRIPS[t][1], l3 = TRIPS[t][2];
    int n1 = 2*l1+1, n2 = 2*l2+1, n3 = 2*l3+1;
    const float* cp = cg + TRIP_OFF[t];
    int o1 = LOFF[l1], o2 = LOFF[l2], o3 = LOFF[l3];
    for (int i = 0; i < n1; ++i) {
      float ai = a[o1 + i];
      for (int j = 0; j < n2; ++j) {
        float ab = ai * b[o2 + j];
        for (int k = 0; k < n3; ++k)
          o[o3 + k] += ab * cp[(i*n2 + j)*n3 + k];
      }
    }
  }
}

__global__ __launch_bounds__(128) void symcontract_kernel(
    const float* __restrict__ X, float* __restrict__ outb, const float* __restrict__ cgg,
    const int* __restrict__ species,
    const float* s10, const float* s20, const float* s30,
    const float* s11, const float* s21, const float* s31, int withL1) {
  __shared__ float cg[CGN];
  for (int t = threadIdx.x; t < CGN; t += 128) cg[t] = cgg[t];
  __syncthreads();
  int n = blockIdx.x, c = threadIdx.x, sp = species[n];
  float x[16], c2[16], c3[16], o[16];
  for (int s = 0; s < 16; ++s) { x[s] = X[((size_t)n*16 + s)*128 + c]; c2[s] = 0.f; c3[s] = 0.f; o[s] = 0.f; }
  cg_contract(x, x, c2, cg);
  cg_contract(c2, x, c3, cg);
  size_t si = (size_t)sp*128 + c;
  o[0] = s10[si]*x[0] + s20[si]*c2[0] + s30[si]*c3[0];
  if (withL1)
    for (int m = 0; m < 3; ++m)
      o[1+m] = s11[si]*x[1+m] + s21[si]*c2[1+m] + s31[si]*c3[1+m];
  for (int s = 0; s < 16; ++s) outb[((size_t)n*16 + s)*128 + c] = o[s];
}

// ---------------- readouts ----------------
__global__ __launch_bounds__(128) void readout_dot_kernel(
    const float* __restrict__ f, const float* __restrict__ w, float* __restrict__ Es, float scale) {
  int n = blockIdx.x, c = threadIdx.x;
  __shared__ float red[128];
  red[c] = f[((size_t)n*16)*128 + c] * w[c];
  __syncthreads();
  for (int s = 64; s; s >>= 1) { if (c < s) red[c] += red[c + s]; __syncthreads(); }
  if (c == 0) Es[n] += red[0] * scale;
}

__global__ __launch_bounds__(128) void readout_mlp_kernel(
    const float* __restrict__ f, const float* __restrict__ Wm, const float* __restrict__ wr,
    float* __restrict__ Es, float rsC) {
  int n = blockIdx.x, c = threadIdx.x;
  __shared__ float fv[128];
  __shared__ float hv[16];
  fv[c] = f[((size_t)n*16)*128 + c];
  __syncthreads();
  if (c < 16) {
    float a = 0.f;
    for (int cc = 0; cc < 128; ++cc) a += fv[cc] * Wm[(size_t)cc*16 + c];
    a *= rsC;
    hv[c] = a * (1.f / (1.f + __expf(-a)));
  }
  __syncthreads();
  if (c == 0) {
    float a = 0.f;
    for (int t = 0; t < 16; ++t) a += hv[t] * wr[t];
    Es[n] += a * 0.25f;
  }
}

__global__ void finalize_E_kernel(const float* __restrict__ Es, const float* __restrict__ offs,
    const int* __restrict__ species, const int* __restrict__ gidx, float* __restrict__ Eout, int NN) {
  int n = blockIdx.x * 256 + threadIdx.x;
  if (n >= NN) return;
  float Ei = Es[n] + offs[species[n]];
  atomicAdd(&Eout[gidx[n]], Ei);
}

// ---------------- forces ----------------
__global__ __launch_bounds__(256) void force_kernel(
    const float* __restrict__ vecs, const float* __restrict__ gY, const float* __restrict__ gs,
    const float* __restrict__ cg, const int* __restrict__ senders, const int* __restrict__ receivers,
    float* __restrict__ Fout, int NE) {
  int e = blockIdx.x * 256 + threadIdx.x;
  if (e >= NE) return;
  float vx = vecs[(size_t)e*3+0], vy = vecs[(size_t)e*3+1], vz = vecs[(size_t)e*3+2];
  float r = sqrtf(vx*vx + vy*vy + vz*vz);
  float ir = 1.f / r;
  float u[3] = { vx*ir, vy*ir, vz*ir };
  float xs = r*0.2f, xs2 = xs*xs, xp = xs2*xs2*xs;
  float env = 1.f - 21.f*xp + 35.f*xp*xs - 15.f*xp*xs2;
  float denv = (-105.f*xs2*xs2 + 210.f*xp - 105.f*xp*xs) * 0.2f;
  if (r >= 5.f) { env = 0.f; denv = 0.f; }
  float drb = 0.f;
  for (int j = 0; j < 8; ++j) {
    float b = (float)(j+1) * 0.6283185307f;
    float s = sinf(b*r), cc = cosf(b*r);
    drb += 0.632455532f * (((b*cc*r - s)*ir*ir)*env + s*ir*denv);
  }
  float F[3];
  float g0 = gs[e];
  for (int a = 0; a < 3; ++a) F[a] = g0 * drb * u[a];
  float dU[3][3];
  for (int a = 0; a < 3; ++a)
    for (int b = 0; b < 3; ++b) dU[a][b] = (((a == b) ? 1.f : 0.f) - u[a]*u[b]) * ir;
  const float* gy = gY + (size_t)e*16;
  for (int a = 0; a < 3; ++a)
    for (int b = 0; b < 3; ++b) F[b] += gy[1+a] * dU[a][b];
  const float* c2 = cg + 102;
  const float* c3 = cg + 472;
  float Y2[5]; float dY2[5][3];
  for (int k = 0; k < 5; ++k) { Y2[k] = 0.f; dY2[k][0] = dY2[k][1] = dY2[k][2] = 0.f; }
  for (int k = 0; k < 5; ++k)
    for (int i = 0; i < 3; ++i)
      for (int j = 0; j < 3; ++j) {
        float cv = c2[(i*3+j)*5 + k];
        Y2[k] += u[i]*u[j]*cv;
        for (int b = 0; b < 3; ++b) dY2[k][b] += cv * (dU[i][b]*u[j] + u[i]*dU[j][b]);
      }
  for (int k = 0; k < 5; ++k)
    for (int b = 0; b < 3; ++b) F[b] += gy[4+k] * dY2[k][b];
  for (int k = 0; k < 7; ++k)
    for (int b = 0; b < 3; ++b) {
      float acc = 0.f;
      for (int i = 0; i < 5; ++i)
        for (int j = 0; j < 3; ++j) {
          float cv = c3[(i*3+j)*7 + k];
          acc += cv * (dY2[i][b]*u[j] + Y2[i]*dU[j][b]);
        }
      F[b] += gy[9+k] * acc;
    }
  int sn = senders[e], rc = receivers[e];
  for (int b = 0; b < 3; ++b) {
    atomicAdd(&Fout[(size_t)sn*3 + b],  F[b]);
    atomicAdd(&Fout[(size_t)rc*3 + b], -F[b]);
  }
}

// ---------------- host orchestration ----------------
static inline void gemm(hipStream_t s, const float* A, const float* B, float* C,
                        int M, int N, int K, float scale, int act, int accum,
                        int rpg, int gstr, int roff) {
  dim3 g((N + 63) / 64, (M + 127) / 128), b(256);
  gemm_wmma_kernel<<<g, b, 0, s>>>(A, B, C, M, N, K, scale, act, accum, rpg, gstr, roff);
}

extern "C" void kernel_launch(void* const* d_in, const int* in_sizes, int n_in,
                              void* d_out, int out_size, void* d_ws, size_t ws_size,
                              hipStream_t stream) {
  const float* vecs     = (const float*)d_in[0];
  const int*   species  = (const int*)d_in[1];
  const int*   senders  = (const int*)d_in[2];
  const int*   receivers= (const int*)d_in[3];
  const int*   gidx     = (const int*)d_in[4];
  const int NE = in_sizes[0] / 3;
  const int NN = in_sizes[1];
  const int NG = out_size - 3 * NN;
  auto PP = [&](int i) -> const float* { return (const float*)d_in[6 + i]; };
  // param leaves in jax pytree (sorted-key) order
  const float* emb        = PP(0);
  const float* L0_down[4] = { PP(1), PP(2), PP(3), PP(4) };
  const float* L0_post[2] = { PP(5), PP(6) };
  const float* L0_up0     = PP(7);
  const float* L0_rad[4]  = { PP(8), PP(9), PP(10), PP(11) };
  const float* L0_ro      = PP(12);
  const float* L0_sc[6]   = { PP(13), PP(14), PP(15), PP(16), PP(17), PP(18) }; // 1_0,1_1,2_0,2_1,3_0,3_1
  const float* L0_skipf[4]= { PP(19), PP(20), PP(21), PP(22) };
  const float* L1_down[4] = { PP(23), PP(24), PP(25), PP(26) };
  const float* L1_post0   = PP(27);
  const float* L1_up[2]   = { PP(28), PP(29) };
  const float* L1_rad[4]  = { PP(30), PP(31), PP(32), PP(33) };
  const float* L1_ro      = PP(34);
  const float* L1_rmlp    = PP(35);
  const float* L1_sc[3]   = { PP(36), PP(37), PP(38) };
  const float* L1_skip0   = PP(39);
  const float* offs       = PP(40);

  float* W = (float*)d_ws;
  size_t o = 0;
  auto alloc = [&](size_t n) {                 // 128-byte aligned blocks (float4-safe)
    size_t r = o; o += (n + 31) & ~(size_t)31; return r;
  };
  float* cg   = W + alloc(CGN);
  float* rb   = W + alloc((size_t)NE * 8);
  float* Yb   = W + alloc((size_t)NE * 16);
  float* fA   = W + alloc((size_t)NN * 16 * 128);
  float* fB   = W + alloc((size_t)NN * 16 * 128);
  float* U    = W + alloc((size_t)NN * 16 * 128);
  float* nf   = W + alloc((size_t)NN * 16 * 128);
  float* nd   = W + alloc((size_t)NN * 16 * 128);
  float* nd2  = W + alloc((size_t)NN * 16 * 128);
  float* scout= W + alloc((size_t)NN * 16 * 128);
  float* gY   = W + alloc((size_t)NE * 16);
  float* gs   = W + alloc((size_t)NE);
  float* Es   = W + alloc((size_t)NN);
  int CE = 256;
  {
    const int cands[7] = { 16384, 8192, 4096, 2048, 1024, 512, 256 };
    for (int ci = 0; ci < 7; ++ci)
      if ((o + (size_t)cands[ci] * (3*64 + 1280) + 256) * 4 <= ws_size) { CE = cands[ci]; break; }
  }
  float* h1 = W + alloc((size_t)CE * 64);
  float* h2 = W + alloc((size_t)CE * 64);
  float* h3 = W + alloc((size_t)CE * 64);
  float* wb = W + alloc((size_t)CE * 1280);

  auto zero = [&](float* p, size_t n) {
    zero_kernel<<<(unsigned)((n + 255) / 256), 256, 0, stream>>>(p, n);
  };
  const int LOFFH[4] = { 0, 1, 4, 9 };
  const float rsC = 0.08838834764831845f, rs8 = 0.3535533905932738f;
  const float rs64 = 0.125f, rs8192 = 0.011048543456039806f;
  const float epsC = 0.1f * rsC;

  zero((float*)d_out, (size_t)out_size);
  zero(gY, (size_t)NE * 16); zero(gs, (size_t)NE); zero(Es, (size_t)NN);
  cg_init_kernel<<<1, 32, 0, stream>>>(cg);
  edge_geom_kernel<<<(NE + 255) / 256, 256, 0, stream>>>(vecs, rb, Yb, cg, NE);
  init_feats_kernel<<<NN, 128, 0, stream>>>(fA, emb, species, NN);

  for (int layer = 0; layer < 2; ++layer) {
    // linear_up: feats -> U (per-l 128x128 WMMA GEMM)
    if (layer == 0) {
      gemm(stream, fA, L0_up0, U, NN, 128, 128, rsC, 0, 0, 1, 16, 0);
    } else {
      gemm(stream, fA, L1_up[0], U, NN,     128, 128, rsC, 0, 0, 1, 16, 0);
      gemm(stream, fA, L1_up[1], U, 3 * NN, 128, 128, rsC, 0, 0, 3, 16, 1);
    }
    zero(nf, (size_t)NN * 16 * 128);
    const float* const* rad = layer ? L1_rad : L0_rad;
    int P = layer ? 10 : 4;
    int wstride = P * 128;
    for (int e0 = 0; e0 < NE; e0 += CE) {
      int ce = (NE - e0 < CE) ? (NE - e0) : CE;
      gemm(stream, rb + (size_t)e0 * 8, rad[0], h1, ce, 64, 8,  rs8,  1, 0, 0, 0, 0);
      gemm(stream, h1, rad[1], h2, ce, 64, 64,      rs64, 1, 0, 0, 0, 0);
      gemm(stream, h2, rad[2], h3, ce, 64, 64,      rs64, 1, 0, 0, 0, 0);
      gemm(stream, h3, rad[3], wb, ce, wstride, 64, rs64, 0, 0, 0, 0, 0);
      tp_scatter_kernel<<<ce, 128, 0, stream>>>(U, Yb, wb, cg, senders, receivers,
                                                nf, gY, gs, e0, ce, layer, wstride);
    }
    const float* const* dwn = layer ? L1_down : L0_down;
    for (int l = 0; l < 4; ++l)
      gemm(stream, nf, dwn[l], nd, (2*l+1) * NN, 128, 128, epsC, 0, 0, 2*l+1, 16, LOFFH[l]);
    const float* X = nd;
    if (layer == 0) {
      for (int l = 0; l < 4; ++l)
        species_mm_kernel<<<NN * (2*l+1), 128, 0, stream>>>(nd, nd2, L0_skipf[l], species,
                                                            rs8192, LOFFH[l], 2*l+1, 0);
      X = nd2;
    }
    if (layer == 0)
      symcontract_kernel<<<NN, 128, 0, stream>>>(X, scout, cg, species,
          L0_sc[0], L0_sc[2], L0_sc[4], L0_sc[1], L0_sc[3], L0_sc[5], 1);
    else
      symcontract_kernel<<<NN, 128, 0, stream>>>(X, scout, cg, species,
          L1_sc[0], L1_sc[1], L1_sc[2], nullptr, nullptr, nullptr, 0);
    if (layer == 0) {
      gemm(stream, scout, L0_post[0], fB, NN,     128, 128, rsC, 0, 0, 1, 16, 0);
      gemm(stream, scout, L0_post[1], fB, 3 * NN, 128, 128, rsC, 0, 0, 3, 16, 1);
      readout_dot_kernel<<<NN, 128, 0, stream>>>(fB, L0_ro, Es, rsC);
    } else {
      gemm(stream, scout, L1_post0, fB, NN, 128, 128, rsC, 0, 0, 1, 16, 0);
      species_mm_kernel<<<NN, 128, 0, stream>>>(fA, fB, L1_skip0, species, rs8192, 0, 1, 1);
      readout_mlp_kernel<<<NN, 128, 0, stream>>>(fB, L1_rmlp, L1_ro, Es, rsC);
    }
    float* t = fA; fA = fB; fB = t;
  }
  float* Eout = (float*)d_out;
  float* Fout = Eout + NG;
  finalize_E_kernel<<<(NN + 255) / 256, 256, 0, stream>>>(Es, offs, species, gidx, Eout, NN);
  force_kernel<<<(NE + 255) / 256, 256, 0, stream>>>(vecs, gY, gs, cg, senders, receivers, Fout, NE);
}
